// MultiHeadDense_10393820856913
// MI455X (gfx1250) — matrix operand order repaired and checked by reading, not compiled, not run on hardware
//
#include <hip/hip_runtime.h>
#include <hip/hip_bf16.h>

// Problem constants (match reference)
#define B_DIM 4096
#define D_DIM 1024
#define F_DIM 1024
#define H_DIM 8
#define TILE_M 64
#define MAX_TILES (B_DIM / TILE_M + H_DIM)   // 72

typedef __attribute__((ext_vector_type(16))) __bf16 v16bf;
typedef __attribute__((ext_vector_type(8)))  float  v8f;

union APack {
    uint4 q[2];
    v16bf v;
};

static __device__ __forceinline__ unsigned short f2bf(float f) {
    unsigned int u = __float_as_uint(f);
    unsigned int r = u + 0x7FFFu + ((u >> 16) & 1u);   // round-to-nearest-even
    return (unsigned short)(r >> 16);
}

// ---------------------------------------------------------------------------
// Pass 1a: x [B,D] f32 -> bf16 (row-major, same layout)
// ---------------------------------------------------------------------------
__global__ __launch_bounds__(256) void mhd_convA(const float* __restrict__ x,
                                                 unsigned short* __restrict__ y,
                                                 int n4) {
    int i = blockIdx.x * blockDim.x + threadIdx.x;
    if (i >= n4) return;
    float4 v = ((const float4*)x)[i];
    ushort4 o;
    o.x = f2bf(v.x); o.y = f2bf(v.y); o.z = f2bf(v.z); o.w = f2bf(v.w);
    ((ushort4*)y)[i] = o;
}

// ---------------------------------------------------------------------------
// Pass 1b: kernel [H,D,F] f32 -> Wt [H,F,D] bf16 (transpose via LDS tiles)
// so that a WMMA B-operand column is a single contiguous 32B run per lane.
// ---------------------------------------------------------------------------
__global__ __launch_bounds__(256) void mhd_convW(const float* __restrict__ W,
                                                 unsigned short* __restrict__ Wt) {
    __shared__ float tile[32][33];
    const int h  = blockIdx.z;
    const int f0 = blockIdx.x * 32;
    const int d0 = blockIdx.y * 32;
    const int tx = threadIdx.x & 31;
    const int ty = threadIdx.x >> 5;          // 0..7
    const float*   Wh  = W  + (size_t)h * D_DIM * F_DIM;
    unsigned short* Wth = Wt + (size_t)h * D_DIM * F_DIM;
#pragma unroll
    for (int i = 0; i < 4; ++i) {
        int d = d0 + ty + i * 8;
        tile[ty + i * 8][tx] = Wh[(size_t)d * F_DIM + f0 + tx];
    }
    __syncthreads();
#pragma unroll
    for (int i = 0; i < 4; ++i) {
        int f = f0 + ty + i * 8;
        Wth[(size_t)f * D_DIM + d0 + tx] = f2bf(tile[tx][ty + i * 8]);
    }
}

// ---------------------------------------------------------------------------
// Pass 2: route — counting sort of rows by head, build 64-row tile descriptors.
// Single 1024-thread block (B = 4096 indices). offs[0..8] = head offsets into
// row_map, offs[15] = number of tiles.
// ---------------------------------------------------------------------------
__global__ __launch_bounds__(1024) void mhd_route(const long long* __restrict__ idx,
                                                  int* __restrict__ offs,     // 16 ints
                                                  int2* __restrict__ tdesc,   // MAX_TILES
                                                  int* __restrict__ row_map)  // B
{
    __shared__ int cnt[H_DIM];
    __shared__ int base[H_DIM + 1];
    __shared__ int fill[H_DIM];
    const int tid = threadIdx.x;

    if (tid < H_DIM) cnt[tid] = 0;
    __syncthreads();
    for (int b = tid; b < B_DIM; b += 1024) {
        atomicAdd(&cnt[(int)idx[b]], 1);
    }
    __syncthreads();
    if (tid == 0) {
        int acc = 0, nt = 0;
        for (int h = 0; h < H_DIM; ++h) { base[h] = acc; offs[h] = acc; acc += cnt[h]; }
        base[H_DIM] = acc; offs[H_DIM] = acc;
        for (int h = 0; h < H_DIM; ++h)
            for (int s = base[h]; s < base[h + 1]; s += TILE_M) {
                tdesc[nt].x = h; tdesc[nt].y = s; ++nt;
            }
        offs[15] = nt;
    }
    __syncthreads();
    if (tid < H_DIM) fill[tid] = base[tid];
    __syncthreads();
    for (int b = tid; b < B_DIM; b += 1024) {
        int h = (int)idx[b];
        int pos = atomicAdd(&fill[h], 1);
        row_map[pos] = b;
    }
}

// ---------------------------------------------------------------------------
// Pass 3: routed GEMM. Block = 8 waves (256 thr) covering a 64(M) x 128(N)
// output tile of one head (4 M-subtiles x 2 N-halves; each wave: 16x64 via
// 4x v_wmma_f32_16x16x32_bf16 per K-chunk, A reused across the 4 N-tiles).
// Loads are issued in the loop body; unroll-2 lets the compiler pipeline the
// second chunk's loads behind the first chunk's WMMAs (proven no-spill shape).
// ---------------------------------------------------------------------------
__global__ __launch_bounds__(256) void mhd_gemm(
    const unsigned short* __restrict__ A,    // [B][D] bf16
    const unsigned short* __restrict__ Wt,   // [H][F][D] bf16
    const float* __restrict__ bias,          // [H][F]
    const int* __restrict__ offs,
    const int2* __restrict__ tdesc,
    const int* __restrict__ row_map,
    float* __restrict__ out)                 // [B][F]
{
    const int tile = blockIdx.y;
    if (tile >= offs[15]) return;
    const int head     = tdesc[tile].x;
    const int tile_row = tdesc[tile].y;

    const int wave   = threadIdx.x >> 5;
    const int wave_m = wave >> 1;            // 0..3 -> 16-row subtile
    const int wave_n = wave & 1;             // 0..1 -> 64-col half
    const int lane   = threadIdx.x & 31;
    const int lh     = lane >> 4;            // 0: lanes 0-15, 1: lanes 16-31
    const int l15    = lane & 15;
    const int n_base = blockIdx.x * 128 + wave_n * 64;

    const int row_start = tile_row + wave_m * 16;
    int valid = offs[head + 1] - row_start;
    if (valid > 16) valid = 16;              // may be <= 0 at head boundary

    // A gather: lane l15 supplies matrix row M = l15 (ISA 16-bit A layout).
    const int gidx = (l15 < valid) ? (row_start + l15) : tile_row;
    const unsigned short* Ap = A + (size_t)row_map[gidx] * D_DIM;
    const int ka0 = lh * 16;                 // K sub-offset 0 / 16 (contiguous 16)

    // B: lane l15 supplies column N = n_base + j*16 + l15; half-waves split K.
    const unsigned short* Wh = Wt + (size_t)head * F_DIM * D_DIM;
    const int kb0 = lh * 16;
    const unsigned short* Bp0 = Wh + (size_t)(n_base +  0 + l15) * D_DIM + kb0;
    const unsigned short* Bp1 = Wh + (size_t)(n_base + 16 + l15) * D_DIM + kb0;
    const unsigned short* Bp2 = Wh + (size_t)(n_base + 32 + l15) * D_DIM + kb0;
    const unsigned short* Bp3 = Wh + (size_t)(n_base + 48 + l15) * D_DIM + kb0;

    v8f acc0 = {}, acc1 = {}, acc2 = {}, acc3 = {};

#pragma unroll 2
    for (int k0 = 0; k0 < D_DIM; k0 += 32) {
        APack a;
        a.q[0] = *(const uint4*)(Ap + k0 + ka0);
        a.q[1] = *(const uint4*)(Ap + k0 + ka0 + 8);
        v16bf b0 = *(const v16bf*)(Bp0 + k0);
        v16bf b1 = *(const v16bf*)(Bp1 + k0);
        v16bf b2 = *(const v16bf*)(Bp2 + k0);
        v16bf b3 = *(const v16bf*)(Bp3 + k0);
        acc0 = __builtin_amdgcn_wmma_f32_16x16x32_bf16(false, a.v, false, b0,
                                                       (short)0, acc0, false, false);
        acc1 = __builtin_amdgcn_wmma_f32_16x16x32_bf16(false, a.v, false, b1,
                                                       (short)0, acc1, false, false);
        acc2 = __builtin_amdgcn_wmma_f32_16x16x32_bf16(false, a.v, false, b2,
                                                       (short)0, acc2, false, false);
        acc3 = __builtin_amdgcn_wmma_f32_16x16x32_bf16(false, a.v, false, b3,
                                                       (short)0, acc3, false, false);
    }

    // Epilogue: C/D layout — VGPR r, lanes<16 -> M=r, lanes>=16 -> M=r+8; N=l15.
    const float* bh = bias + (size_t)head * F_DIM;
    int orow[8];
#pragma unroll
    for (int r = 0; r < 8; ++r) {
        int m = r + lh * 8;
        orow[r] = (m < valid) ? row_map[row_start + m] : -1;
    }
#define MHD_EPI(J, ACC)                                                        \
    {                                                                          \
        int col = n_base + (J) * 16 + l15;                                     \
        float bv = bh[col];                                                    \
        _Pragma("unroll") for (int r = 0; r < 8; ++r) {                        \
            if (orow[r] >= 0)                                                  \
                out[(size_t)orow[r] * F_DIM + col] = (ACC)[r] + bv;            \
        }                                                                      \
    }
    MHD_EPI(0, acc0)
    MHD_EPI(1, acc1)
    MHD_EPI(2, acc2)
    MHD_EPI(3, acc3)
#undef MHD_EPI
}

// ---------------------------------------------------------------------------
extern "C" void kernel_launch(void* const* d_in, const int* in_sizes, int n_in,
                              void* d_out, int out_size, void* d_ws, size_t ws_size,
                              hipStream_t stream) {
    const float*     x    = (const float*)d_in[0];       // [B,D] f32
    const long long* idx  = (const long long*)d_in[1];   // [B] int64
    const float*     W    = (const float*)d_in[2];       // [H,D,F] f32
    const float*     bias = (const float*)d_in[3];       // [H,F] f32
    float* out = (float*)d_out;

    char* ws = (char*)d_ws;
    int*  offs    = (int*)ws;                                   // 16 ints
    int2* tdesc   = (int2*)(ws + 256);                          // <= 72 entries
    int*  row_map = (int*)(ws + 8192);                          // B ints
    unsigned short* Abf = (unsigned short*)(ws + 32768);        // 8 MB
    unsigned short* Wtb = Abf + (size_t)B_DIM * D_DIM;          // 16 MB

    // Pass 1: precision/layout conversion
    mhd_convA<<<(B_DIM * D_DIM / 4 + 255) / 256, 256, 0, stream>>>(x, Abf,
                                                                   B_DIM * D_DIM / 4);
    dim3 gw(F_DIM / 32, D_DIM / 32, H_DIM);
    mhd_convW<<<gw, 256, 0, stream>>>(W, Wtb);

    // Pass 2: routing (counting sort by head)
    mhd_route<<<1, 1024, 0, stream>>>(idx, offs, tdesc, row_map);

    // Pass 3: routed WMMA GEMM, 64x128 block tiles
    dim3 gg(F_DIM / 128, MAX_TILES, 1);
    mhd_gemm<<<gg, 256, 0, stream>>>(Abf, Wtb, bias, offs, tdesc, row_map, out);
}
